// SpatialAttentionBlock_6012954214529
// MI455X (gfx1250) — compile-verified, matrix operand order checked by reading, not run on hardware
//
#include <hip/hip_runtime.h>
#include <hip/hip_bf16.h>

// ---------------- constants ----------------
#define Bb 64
#define Tt 128
#define Vv 25
#define Dd 256
#define Hh 8
#define HDd 32
#define NROWS (Bb * Tt * Vv)   // 204800 token rows
#define NSEQ  (Bb * Tt)        // 8192 sequences

typedef __attribute__((ext_vector_type(16))) _Float16 v16h;
typedef __attribute__((ext_vector_type(8)))  float    v8f;

union F16x16 { v16h v; unsigned int u[8]; };

// Load a 16x32 f16 WMMA fragment (A from row-major [M][K]; B from K-contiguous wT [N][K]).
// ISA layout: lanes 0-15 -> rows, K-halves 0..7 & 16..23; lanes 16-31 -> K 8..15 & 24..31.
__device__ inline v16h frag_ld(const _Float16* base, int ld, int r0, int k0) {
  int lane = threadIdx.x & 31;
  const unsigned int* q =
      (const unsigned int*)(base + (size_t)(r0 + (lane & 15)) * ld + k0 + ((lane >> 4) << 3));
  F16x16 f;
#pragma unroll
  for (int i = 0; i < 4; ++i) { f.u[i] = q[i]; f.u[4 + i] = q[8 + i]; }
  return f.v;
}

__device__ inline v8f wmma16(v16h a, v16h b, v8f c) {
  return __builtin_amdgcn_wmma_f32_16x16x32_f16(false, a, false, b, (short)0, c, false, false);
}

// ---------------- K0: weight convert + transpose (fp32 [K][N] -> f16 [N][K]) --------
__global__ __launch_bounds__(256) void k_convert_wT(const float* __restrict__ w,
                                                    _Float16* __restrict__ wT,
                                                    int K, int Nout) {
  int idx = blockIdx.x * 256 + threadIdx.x;
  if (idx >= K * Nout) return;
  int n = idx / K, k = idx - n * K;
  wT[idx] = (_Float16)w[(size_t)k * Nout + n];
}

// ---------------- K1: LN1 + QKV GEMM (M=32 tile, N=768, K=256) ----------------
__global__ __launch_bounds__(256) void k_ln1_qkv(const float* __restrict__ x,
                                                 const float* __restrict__ g,
                                                 const float* __restrict__ bln,
                                                 const _Float16* __restrict__ wT,
                                                 const float* __restrict__ bias,
                                                 _Float16* __restrict__ qkv_out) {
  __shared__ _Float16 aLds[32 * 264];
  int tid = threadIdx.x;
  int row0 = blockIdx.x * 32;
  // LayerNorm: 32 rows x (8 lanes x 32 elems)
  int r = tid >> 3, c = (tid & 7) * 32;
  const float* xr = x + (size_t)(row0 + r) * Dd + c;
  float vals[32];
  float s = 0.f, s2 = 0.f;
#pragma unroll
  for (int i = 0; i < 32; ++i) { float v = xr[i]; vals[i] = v; s += v; }
#pragma unroll
  for (int m = 4; m >= 1; m >>= 1) s += __shfl_xor(s, m);
  float mean = s * (1.0f / Dd);
#pragma unroll
  for (int i = 0; i < 32; ++i) { float d = vals[i] - mean; s2 += d * d; }
#pragma unroll
  for (int m = 4; m >= 1; m >>= 1) s2 += __shfl_xor(s2, m);
  float inv = rsqrtf(s2 * (1.0f / Dd) + 1e-5f);
#pragma unroll
  for (int i = 0; i < 32; ++i)
    aLds[r * 264 + c + i] = (_Float16)((vals[i] - mean) * inv * g[c + i] + bln[c + i]);
  __syncthreads();

  int wave = tid >> 5, lane = tid & 31;
  int n_base = wave * 96;                       // 8 waves x 96 cols = 768
  v8f acc0[6] = {}, acc1[6] = {};
  for (int k0 = 0; k0 < 256; k0 += 32) {
    v16h a0 = frag_ld(aLds, 264, 0, k0);
    v16h a1 = frag_ld(aLds, 264, 16, k0);
    if (k0 + 32 < 256)
      __builtin_prefetch(wT + (size_t)n_base * 256 + k0 + 32, 0, 3);
#pragma unroll
    for (int t = 0; t < 6; ++t) {
      v16h b = frag_ld(wT, 256, n_base + t * 16, k0);   // B fetched once, used twice
      acc0[t] = wmma16(a0, b, acc0[t]);
      acc1[t] = wmma16(a1, b, acc1[t]);
    }
  }
  int nlo = lane & 15, mhi = (lane >> 4) << 3;
#pragma unroll
  for (int t = 0; t < 6; ++t) {
    int n = n_base + t * 16 + nlo;
    float bv = bias[n];
#pragma unroll
    for (int rr = 0; rr < 8; ++rr) {
      qkv_out[(size_t)(row0 + mhi + rr) * 768 + n]      = (_Float16)(acc0[t][rr] + bv);
      qkv_out[(size_t)(row0 + 16 + mhi + rr) * 768 + n] = (_Float16)(acc1[t][rr] + bv);
    }
  }
}

// ---------------- K2: cosine attention, one wave per head, 4 heads/block ----------
__global__ __launch_bounds__(128) void k_attn(const _Float16* __restrict__ qkv,
                                              const float* __restrict__ logit_scale,
                                              _Float16* __restrict__ att_out) {
  __shared__ _Float16 qf[4][32 * 40];   // q, later reused for attn probs
  __shared__ _Float16 kf[4][32 * 40];   // k rows [j][d]
  __shared__ _Float16 vT[4][32 * 40];   // v transposed [d][j]
  __shared__ float    sm[4][32 * 33];   // f32 logits
  int tid = threadIdx.x;
  int lane = tid & 31, w = tid >> 5;
  int n = blockIdx.x >> 1;
  int h = ((blockIdx.x & 1) << 2) + w;
  _Float16* q_s = qf[w]; _Float16* k_s = kf[w]; _Float16* v_s = vT[w]; float* s_s = sm[w];

  for (int i = lane; i < 32 * 40; i += 32) {
    q_s[i] = (_Float16)0.f; k_s[i] = (_Float16)0.f; v_s[i] = (_Float16)0.f;
  }
  const _Float16* base = qkv + (size_t)n * Vv * 768 + h * HDd;
  for (int rr = 0; rr < Vv; ++rr) {
    q_s[rr * 40 + lane] = base[rr * 768 + lane];
    k_s[rr * 40 + lane] = base[rr * 768 + 256 + lane];
    v_s[lane * 40 + rr] = base[rr * 768 + 512 + lane];
  }
  __syncthreads();

  // row-wise L2 normalize q,k; fold temperature and 1/sqrt(d) into q
  float sc = fminf(expf(logit_scale[h]), 100.f) * 0.17677669529663687f; // /sqrt(32)
  if (lane < Vv) {
    float sq = 0.f, sk = 0.f;
    for (int d = 0; d < 32; ++d) {
      float qv = (float)q_s[lane * 40 + d], kv = (float)k_s[lane * 40 + d];
      sq += qv * qv; sk += kv * kv;
    }
    float iq = sc / fmaxf(sqrtf(sq), 1e-12f);
    float ik = 1.f / fmaxf(sqrtf(sk), 1e-12f);
    for (int d = 0; d < 32; ++d) {
      q_s[lane * 40 + d] = (_Float16)((float)q_s[lane * 40 + d] * iq);
      k_s[lane * 40 + d] = (_Float16)((float)k_s[lane * 40 + d] * ik);
    }
  }
  __syncthreads();

  // logits = q (32x32) x k^T : A=q rows, B=k rows treated as [n=j][k=d]
  v8f c00 = {}, c01 = {}, c10 = {}, c11 = {};
  {
    v16h a0 = frag_ld(q_s, 40, 0, 0), a1 = frag_ld(q_s, 40, 16, 0);
    v16h b0 = frag_ld(k_s, 40, 0, 0), b1 = frag_ld(k_s, 40, 16, 0);
    c00 = wmma16(a0, b0, c00); c01 = wmma16(a0, b1, c01);
    c10 = wmma16(a1, b0, c10); c11 = wmma16(a1, b1, c11);
  }
  int nlo = lane & 15, mhi = (lane >> 4) << 3;
#pragma unroll
  for (int rr = 0; rr < 8; ++rr) {
    int m = mhi + rr;
    s_s[m * 33 + nlo]             = c00[rr];
    s_s[m * 33 + 16 + nlo]        = c01[rr];
    s_s[(16 + m) * 33 + nlo]      = c10[rr];
    s_s[(16 + m) * 33 + 16 + nlo] = c11[rr];
  }
  __syncthreads();

  // softmax over j<25; write probs (f16) into q_s (q no longer needed)
  {
    int r2 = lane;
    if (r2 < Vv) {
      float mx = -1e30f;
      for (int j = 0; j < Vv; ++j) mx = fmaxf(mx, s_s[r2 * 33 + j]);
      float e[Vv]; float sum = 0.f;
      for (int j = 0; j < Vv; ++j) { float ev = expf(s_s[r2 * 33 + j] - mx); e[j] = ev; sum += ev; }
      float is = 1.f / sum;
      for (int j = 0; j < Vv; ++j) q_s[r2 * 40 + j] = (_Float16)(e[j] * is);
      for (int j = Vv; j < 32; ++j) q_s[r2 * 40 + j] = (_Float16)0.f;
    } else {
      for (int j = 0; j < 32; ++j) q_s[r2 * 40 + j] = (_Float16)0.f;
    }
  }
  __syncthreads();

  // out = attn (32x32) x v : B = vT [n=d][k=j]
  v8f o00 = {}, o01 = {}, o10 = {}, o11 = {};
  {
    v16h a0 = frag_ld(q_s, 40, 0, 0), a1 = frag_ld(q_s, 40, 16, 0);
    v16h b0 = frag_ld(v_s, 40, 0, 0), b1 = frag_ld(v_s, 40, 16, 0);
    o00 = wmma16(a0, b0, o00); o01 = wmma16(a0, b1, o01);
    o10 = wmma16(a1, b0, o10); o11 = wmma16(a1, b1, o11);
  }
  _Float16* ob = att_out + (size_t)n * Vv * Dd + h * HDd;
#pragma unroll
  for (int rr = 0; rr < 8; ++rr) {
    int m = mhi + rr;
    if (m < Vv) {
      ob[m * Dd + nlo]      = (_Float16)o00[rr];
      ob[m * Dd + 16 + nlo] = (_Float16)o01[rr];
    }
    int m2 = 16 + mhi + rr;
    if (m2 < Vv) {
      ob[m2 * Dd + nlo]      = (_Float16)o10[rr];
      ob[m2 * Dd + 16 + nlo] = (_Float16)o11[rr];
    }
  }
}

// ---------------- K3: proj GEMM + bias + residual (M=32, async LDS staging) -------
__global__ __launch_bounds__(256) void k_proj_res(const _Float16* __restrict__ att,
                                                  const _Float16* __restrict__ wT,
                                                  const float* __restrict__ bias,
                                                  const float* __restrict__ x,
                                                  float* __restrict__ xf) {
  __shared__ _Float16 aLds[32 * 264];
  int tid = threadIdx.x;
  int row0 = blockIdx.x * 32;
  // Stage 32x256 f16 A-tile via CDNA5 async global->LDS (ASYNCcnt), 16B chunks.
#pragma unroll
  for (int i = 0; i < 4; ++i) {
    int chunk = tid + i * 256;                 // 0..1023
    int rr = chunk >> 5, c8 = chunk & 31;
    unsigned lds_off = (unsigned)(uintptr_t)(&aLds[rr * 264 + c8 * 8]);
    unsigned long long gaddr =
        (unsigned long long)(uintptr_t)(att + (size_t)(row0 + rr) * Dd + c8 * 8);
    asm volatile("global_load_async_to_lds_b128 %0, %1, off"
                 :: "v"(lds_off), "v"(gaddr) : "memory");
  }
  asm volatile("s_wait_asynccnt 0x0" ::: "memory");
  __syncthreads();

  int wave = tid >> 5, lane = tid & 31;
  int n_base = wave * 32;                       // 8 waves x 32 cols = 256
  v8f acc0[2] = {}, acc1[2] = {};
  for (int k0 = 0; k0 < 256; k0 += 32) {
    v16h a0 = frag_ld(aLds, 264, 0, k0);
    v16h a1 = frag_ld(aLds, 264, 16, k0);
#pragma unroll
    for (int t = 0; t < 2; ++t) {
      v16h b = frag_ld(wT, 256, n_base + t * 16, k0);
      acc0[t] = wmma16(a0, b, acc0[t]);
      acc1[t] = wmma16(a1, b, acc1[t]);
    }
  }
  int nlo = lane & 15, mhi = (lane >> 4) << 3;
#pragma unroll
  for (int t = 0; t < 2; ++t) {
    int n = n_base + t * 16 + nlo;
    float bv = bias[n];
#pragma unroll
    for (int rr = 0; rr < 8; ++rr) {
      size_t i0 = (size_t)(row0 + mhi + rr) * Dd + n;
      size_t i1 = (size_t)(row0 + 16 + mhi + rr) * Dd + n;
      xf[i0] = acc0[t][rr] + bv + x[i0];
      xf[i1] = acc1[t][rr] + bv + x[i1];
    }
  }
}

// ---------------- K4: LN2 + FFN1 + GeLU + FFN2 + residual (M=32, dynamic LDS) -----
__global__ __launch_bounds__(256) void k_ffn(const float* __restrict__ g2,
                                             const float* __restrict__ b2ln,
                                             const _Float16* __restrict__ w1T,
                                             const float* __restrict__ bias1,
                                             const _Float16* __restrict__ w2T,
                                             const float* __restrict__ bias2,
                                             float* __restrict__ xf) {
  extern __shared__ __align__(16) char smem[];
  _Float16* aLds = (_Float16*)smem;              // 32 x 264
  _Float16* hLds = aLds + 32 * 264;              // 32 x 1032
  int tid = threadIdx.x;
  int row0 = blockIdx.x * 32;
  int r = tid >> 3, c = (tid & 7) * 32;
  // LN2 over 32 rows
  const float* xr = xf + (size_t)(row0 + r) * Dd + c;
  float vals[32];
  float s = 0.f, s2 = 0.f;
#pragma unroll
  for (int i = 0; i < 32; ++i) { float v = xr[i]; vals[i] = v; s += v; }
#pragma unroll
  for (int m = 4; m >= 1; m >>= 1) s += __shfl_xor(s, m);
  float mean = s * (1.0f / Dd);
#pragma unroll
  for (int i = 0; i < 32; ++i) { float d = vals[i] - mean; s2 += d * d; }
#pragma unroll
  for (int m = 4; m >= 1; m >>= 1) s2 += __shfl_xor(s2, m);
  float inv = rsqrtf(s2 * (1.0f / Dd) + 1e-5f);
#pragma unroll
  for (int i = 0; i < 32; ++i)
    aLds[r * 264 + c + i] = (_Float16)((vals[i] - mean) * inv * g2[c + i] + b2ln[c + i]);
  __syncthreads();

  int wave = tid >> 5, lane = tid & 31;
  int nlo = lane & 15, mhi = (lane >> 4) << 3;
  // FFN1: N=1024, each wave covers 128 cols (8 N-tiles) x 2 M-frags
  {
    int n_base = wave * 128;
    v8f acc0[8] = {}, acc1[8] = {};
    for (int k0 = 0; k0 < 256; k0 += 32) {
      v16h a0 = frag_ld(aLds, 264, 0, k0);
      v16h a1 = frag_ld(aLds, 264, 16, k0);
      if (k0 + 32 < 256)
        __builtin_prefetch(w1T + (size_t)n_base * 256 + k0 + 32, 0, 3);
#pragma unroll
      for (int t = 0; t < 8; ++t) {
        v16h b = frag_ld(w1T, 256, n_base + t * 16, k0);
        acc0[t] = wmma16(a0, b, acc0[t]);
        acc1[t] = wmma16(a1, b, acc1[t]);
      }
    }
#pragma unroll
    for (int t = 0; t < 8; ++t) {
      int n = n_base + t * 16 + nlo;
      float bv = bias1[n];
#pragma unroll
      for (int rr = 0; rr < 8; ++rr) {
        float v0 = acc0[t][rr] + bv;
        v0 = 0.5f * v0 * (1.0f + erff(v0 * 0.7071067811865476f)); // exact GeLU
        hLds[(mhi + rr) * 1032 + n] = (_Float16)v0;
        float v1 = acc1[t][rr] + bv;
        v1 = 0.5f * v1 * (1.0f + erff(v1 * 0.7071067811865476f));
        hLds[(16 + mhi + rr) * 1032 + n] = (_Float16)v1;
      }
    }
  }
  __syncthreads();
  // FFN2: K=1024, N=256, each wave covers 32 cols (2 N-tiles) x 2 M-frags
  {
    int n_base = wave * 32;
    v8f acc0[2] = {}, acc1[2] = {};
    for (int k0 = 0; k0 < 1024; k0 += 32) {
      v16h a0 = frag_ld(hLds, 1032, 0, k0);
      v16h a1 = frag_ld(hLds, 1032, 16, k0);
#pragma unroll
      for (int t = 0; t < 2; ++t) {
        v16h b = frag_ld(w2T, 1024, n_base + t * 16, k0);
        acc0[t] = wmma16(a0, b, acc0[t]);
        acc1[t] = wmma16(a1, b, acc1[t]);
      }
    }
#pragma unroll
    for (int t = 0; t < 2; ++t) {
      int n = n_base + t * 16 + nlo;
      float bv = bias2[n];
#pragma unroll
      for (int rr = 0; rr < 8; ++rr) {
        size_t i0 = (size_t)(row0 + mhi + rr) * Dd + n;
        size_t i1 = (size_t)(row0 + 16 + mhi + rr) * Dd + n;
        xf[i0] = acc0[t][rr] + bv + xf[i0];   // residual re-read (L2-hot)
        xf[i1] = acc1[t][rr] + bv + xf[i1];
      }
    }
  }
}

// ---------------- workspace layout (in f16 elements) ----------------
static const size_t OFF_WQKV = 0;
static const size_t OFF_WPROJ = OFF_WQKV + 768 * 256;
static const size_t OFF_WF1   = OFF_WPROJ + 256 * 256;
static const size_t OFF_WF2   = OFF_WF1 + 1024 * 256;
static const size_t OFF_QKV   = OFF_WF2 + 256 * 1024;
static const size_t OFF_ATT   = OFF_QKV + (size_t)NROWS * 768;

extern "C" void kernel_launch(void* const* d_in, const int* in_sizes, int n_in,
                              void* d_out, int out_size, void* d_ws, size_t ws_size,
                              hipStream_t stream) {
  const float* x        = (const float*)d_in[0];
  const float* ln1_g    = (const float*)d_in[1];
  const float* ln1_b    = (const float*)d_in[2];
  const float* qkv_w    = (const float*)d_in[3];
  const float* qkv_b    = (const float*)d_in[4];
  const float* proj_w   = (const float*)d_in[5];
  const float* proj_b   = (const float*)d_in[6];
  const float* lscale   = (const float*)d_in[7];
  const float* ln2_g    = (const float*)d_in[8];
  const float* ln2_b    = (const float*)d_in[9];
  const float* ffn_w1   = (const float*)d_in[10];
  const float* ffn_b1   = (const float*)d_in[11];
  const float* ffn_w2   = (const float*)d_in[12];
  const float* ffn_b2   = (const float*)d_in[13];
  float* out = (float*)d_out;
  _Float16* ws = (_Float16*)d_ws;

  // weights -> f16, transposed to [N][K]
  k_convert_wT<<<768, 256, 0, stream>>>(qkv_w,  ws + OFF_WQKV, 256, 768);
  k_convert_wT<<<256, 256, 0, stream>>>(proj_w, ws + OFF_WPROJ, 256, 256);
  k_convert_wT<<<1024, 256, 0, stream>>>(ffn_w1, ws + OFF_WF1, 256, 1024);
  k_convert_wT<<<1024, 256, 0, stream>>>(ffn_w2, ws + OFF_WF2, 1024, 256);

  k_ln1_qkv<<<NROWS / 32, 256, 0, stream>>>(x, ln1_g, ln1_b, ws + OFF_WQKV, qkv_b,
                                            ws + OFF_QKV);
  k_attn<<<NSEQ * 2, 128, 0, stream>>>(ws + OFF_QKV, lscale, ws + OFF_ATT);
  k_proj_res<<<NROWS / 32, 256, 0, stream>>>(ws + OFF_ATT, ws + OFF_WPROJ, proj_b, x, out);
  // K4 needs 32*264 + 32*1032 f16 = 82944 B dynamic LDS (gfx1250: 320 KB/WG)
  k_ffn<<<NROWS / 32, 256, (32 * 264 + 32 * 1032) * 2, stream>>>(
      ln2_g, ln2_b, ws + OFF_WF1, ffn_b1, ws + OFF_WF2, ffn_b2, out);
}
